// OmniBlock_6004364280335
// MI455X (gfx1250) — matrix operand order repaired and checked by reading, not compiled, not run on hardware
//
#include <hip/hip_runtime.h>
#include <hip/hip_bf16.h>
#include <stdint.h>

// ---------------------------------------------------------------- types
typedef __bf16 bf16;
typedef __attribute__((ext_vector_type(16))) __bf16 bf16x16;
typedef __attribute__((ext_vector_type(8)))  __bf16 bf16x8;
typedef __attribute__((ext_vector_type(8)))  float  f32x8;

union Frag16 { bf16x16 v; bf16x8 h[2]; };

__device__ __forceinline__ bf16 f2bf(float f) {
  unsigned u = __builtin_bit_cast(unsigned, f);
  unsigned r = u + 0x7FFFu + ((u >> 16) & 1u);   // round-to-nearest-even
  return __builtin_bit_cast(bf16, (unsigned short)(r >> 16));
}

// CDNA5 async global->LDS copy (16 bytes per lane), tracked by ASYNCcnt.
__device__ __forceinline__ void async_lds_b128(void* lds, const void* g) {
  asm volatile("global_load_async_to_lds_b128 %0, %1, off"
               :: "v"((uint32_t)(uintptr_t)lds), "v"((uint64_t)(uintptr_t)g)
               : "memory");
}
__device__ __forceinline__ void wait_asynccnt0() {
  asm volatile("s_wait_asynccnt 0x0" ::: "memory");
}

#define H_DIM   1024
#define N_TOK   4096      // B*N = 2*2048
#define SEQ     2048
#define NHEADS  16
#define HDIM    64
#define MAPSTRIDE 4096

// ---------------------------------------------------------------- layernorm -> bf16
__global__ __launch_bounds__(256)
void layernorm_bf16(const float* __restrict__ x, const float* __restrict__ g,
                    const float* __restrict__ be, bf16* __restrict__ out)
{
  __shared__ float redS[8];
  __shared__ float redQ[8];
  const int row  = blockIdx.x;
  const int tid  = threadIdx.x;
  const int wave = tid >> 5, lane = tid & 31;
  const float* xr = x + (size_t)row * H_DIM;

  float v4[4]; float s = 0.f, q = 0.f;
  #pragma unroll
  for (int i = 0; i < 4; ++i) {
    v4[i] = xr[tid + i * 256];
    s += v4[i]; q += v4[i] * v4[i];
  }
  #pragma unroll
  for (int off = 1; off < 32; off <<= 1) {
    s += __shfl_xor(s, off, 32);
    q += __shfl_xor(q, off, 32);
  }
  if (lane == 0) { redS[wave] = s; redQ[wave] = q; }
  __syncthreads();
  float ts = 0.f, tq = 0.f;
  #pragma unroll
  for (int i = 0; i < 8; ++i) { ts += redS[i]; tq += redQ[i]; }
  const float mean = ts * (1.f / H_DIM);
  const float var  = tq * (1.f / H_DIM) - mean * mean;
  const float rstd = rsqrtf(var + 1e-6f);
  #pragma unroll
  for (int i = 0; i < 4; ++i) {
    int c = tid + i * 256;
    out[(size_t)row * H_DIM + c] = f2bf((v4[i] - mean) * rstd * g[c] + be[c]);
  }
}

// ---------------------------------------------------------------- generic WMMA GEMM
// C[M,Ncols] = A(bf16)[M,K] @ W(f32)[K,ldw] (+bias) with optional row gather/scatter.
// mode 0: outF = acc+bias      mode 1: outF = resid+acc+bias
// mode 2: outB = bf16(acc+bias) mode 3: outF += acc+bias
#define TILE_M 128
#define TILE_N 128
#define TILE_K 32

__global__ __launch_bounds__(256)
void gemm_bf16_wmma(const bf16* __restrict__ A, int lda,
                    const int* __restrict__ rowmap, const int* __restrict__ counts,
                    int keyBase, int Mdense,
                    const float* __restrict__ Wbase, size_t wStride, int ldw, int K,
                    const float* __restrict__ bias,
                    float* __restrict__ outF, bf16* __restrict__ outB,
                    const float* __restrict__ resid, int ldo, int mode)
{
  __shared__ __align__(32) bf16 Atile[TILE_M * TILE_K];
  __shared__ __align__(32) bf16 Btile[TILE_N * TILE_K];

  const int z = blockIdx.z;
  const float* W = Wbase + (size_t)z * wStride;
  int Meff = Mdense;
  const int* rmap = nullptr;
  if (rowmap) { Meff = counts[keyBase + z]; rmap = rowmap + (size_t)(keyBase + z) * MAPSTRIDE; }
  const int mtile = blockIdx.y * TILE_M;
  if (mtile >= Meff) return;
  const int ntile = blockIdx.x * TILE_N;

  const int tid  = threadIdx.x;
  const int wave = tid >> 5, lane = tid & 31;
  const int wm = wave & 1, wn = wave >> 1;       // 2 x 4 wave grid
  const int lm = lane & 15, hi = lane >> 4;

  f32x8 acc[4][2];
  #pragma unroll
  for (int i = 0; i < 4; ++i)
    #pragma unroll
    for (int j = 0; j < 2; ++j) acc[i][j] = f32x8{0,0,0,0,0,0,0,0};

  // A staging: 128 rows x 32 k, 16 bf16 per thread, async global->LDS
  const int ar = tid >> 1;
  const int ah = (tid & 1) * 16;
  int arow = mtile + ar; if (arow >= Meff) arow = Meff - 1;
  const int asrc = rmap ? rmap[arow] : arow;
  const bf16* aptr = A + (size_t)asrc * lda + ah;
  bf16* alds = &Atile[ar * TILE_K + ah];
  // B staging: transpose f32 W[K,ldw] -> bf16 Btile[n][k]; loop-carried pointers
  const int bn = tid & 127;
  const int bk = (tid >> 7) * 16;
  const size_t ldwE  = (size_t)ldw;            // element stride between k rows
  const size_t slabE = (size_t)TILE_K * ldwE;  // element stride between k0 slabs
  const float* bptr = W + (size_t)(ntile + bn) + (size_t)bk * ldwE;

  for (int k0 = 0; k0 < K; k0 += TILE_K) {
    async_lds_b128(alds,     aptr + k0);
    async_lds_b128(alds + 8, aptr + k0 + 8);
    {
      const float* p = bptr;
      bf16x16 t;
      #pragma unroll
      for (int kk = 0; kk < 16; ++kk) { t[kk] = f2bf(*p); p += ldwE; }
      *(bf16x16*)&Btile[bn * TILE_K + bk] = t;
    }
    if (k0 + TILE_K < K)
      __builtin_prefetch(bptr + slabE, 0, 1);
    bptr += slabE;
    wait_asynccnt0();
    __syncthreads();

    bf16x16 af[4];
    #pragma unroll
    for (int i = 0; i < 4; ++i) {
      const bf16* p = &Atile[(wm * 64 + i * 16 + lm) * TILE_K];
      Frag16 f;
      f.h[0] = *(const bf16x8*)(p + 16 * hi);
      f.h[1] = *(const bf16x8*)(p + 16 * hi + 8);
      af[i] = f.v;
    }
    bf16x16 bfr[2];
    #pragma unroll
    for (int j = 0; j < 2; ++j)
      bfr[j] = *(const bf16x16*)&Btile[(wn * 32 + j * 16 + lm) * TILE_K + 16 * hi];
    #pragma unroll
    for (int i = 0; i < 4; ++i)
      #pragma unroll
      for (int j = 0; j < 2; ++j)
        acc[i][j] = __builtin_amdgcn_wmma_f32_16x16x32_bf16(
            false, af[i], false, bfr[j], (short)0, acc[i][j], false, false);
    __syncthreads();
  }

  #pragma unroll
  for (int i = 0; i < 4; ++i) {
    #pragma unroll
    for (int j = 0; j < 2; ++j) {
      const int col = ntile + wn * 32 + j * 16 + lm;
      const float bv = bias ? bias[col] : 0.f;
      #pragma unroll
      for (int v = 0; v < 8; ++v) {
        const int lrow = wm * 64 + i * 16 + v + 8 * hi;
        const int grow = mtile + lrow;
        if (grow >= Meff) continue;
        const int orow = rmap ? rmap[grow] : grow;
        const float val = acc[i][j][v] + bv;
        const size_t oidx = (size_t)orow * ldo + col;
        if      (mode == 0) outF[oidx] = val;
        else if (mode == 1) outF[oidx] = resid[oidx] + val;
        else if (mode == 2) outB[oidx] = f2bf(val);
        else                outF[oidx] += val;
      }
    }
  }
}

// ---------------------------------------------------------------- qkv split (+layout) -> bf16
__global__ __launch_bounds__(256)
void split_qkv(const float* __restrict__ qkv, bf16* __restrict__ qb,
               bf16* __restrict__ kb, bf16* __restrict__ vt)
{
  const size_t i = (size_t)blockIdx.x * 256 + threadIdx.x;
  if (i >= (size_t)N_TOK * 3 * H_DIM) return;
  const int t = (int)(i / (3 * H_DIM));
  const int c = (int)(i % (3 * H_DIM));
  const int b = t / SEQ, n = t % SEQ;
  const int which = c / H_DIM;
  const int h = (c % H_DIM) / HDIM, d = c % HDIM;
  const int bh = b * NHEADS + h;
  const bf16 val = f2bf(qkv[i]);
  if (which == 0)      qb[((size_t)bh * SEQ + n) * HDIM + d] = val;
  else if (which == 1) kb[((size_t)bh * SEQ + n) * HDIM + d] = val;
  else                 vt[((size_t)bh * HDIM + d) * SEQ + n] = val;
}

// ---------------------------------------------------------------- flash attention (WMMA)
__global__ __launch_bounds__(256)
void attn_wmma(const bf16* __restrict__ qb, const bf16* __restrict__ kb,
               const bf16* __restrict__ vt, bf16* __restrict__ outb)
{
  __shared__ __align__(32) bf16 Plds[8][16 * 32];
  const int bh   = blockIdx.x;                 // b*16+h
  const int wave = threadIdx.x >> 5, lane = threadIdx.x & 31;
  const int lm = lane & 15, hi = lane >> 4;
  const int qrow0 = (blockIdx.y * 8 + wave) * 16;
  const int b = bh >> 4, h = bh & 15;

  const bf16* qbase = qb + ((size_t)bh * SEQ + qrow0) * HDIM;
  bf16x16 qf[2];
  #pragma unroll
  for (int f = 0; f < 2; ++f) {
    const bf16* p = qbase + (size_t)lm * HDIM + f * 32;
    Frag16 fr;
    fr.h[0] = *(const bf16x8*)(p + 16 * hi);
    fr.h[1] = *(const bf16x8*)(p + 16 * hi + 8);
    qf[f] = fr.v;
  }

  float mrow[8], lrow[8];
  f32x8 o[4];
  #pragma unroll
  for (int v = 0; v < 8; ++v) { mrow[v] = -1e30f; lrow[v] = 0.f; }
  #pragma unroll
  for (int dc = 0; dc < 4; ++dc) o[dc] = f32x8{0,0,0,0,0,0,0,0};

  const bf16* kbase = kb + (size_t)bh * SEQ * HDIM;
  const bf16* vbase = vt + (size_t)bh * HDIM * SEQ;
  bf16* pl = Plds[wave];

  for (int kk = 0; kk < SEQ; kk += 32) {
    f32x8 s0 = f32x8{0,0,0,0,0,0,0,0};
    f32x8 s1 = f32x8{0,0,0,0,0,0,0,0};
    #pragma unroll
    for (int f = 0; f < 2; ++f) {
      bf16x16 bk0 = *(const bf16x16*)(kbase + (size_t)(kk + lm) * HDIM + f * 32 + 16 * hi);
      s0 = __builtin_amdgcn_wmma_f32_16x16x32_bf16(false, qf[f], false, bk0, (short)0, s0, false, false);
      bf16x16 bk1 = *(const bf16x16*)(kbase + (size_t)(kk + 16 + lm) * HDIM + f * 32 + 16 * hi);
      s1 = __builtin_amdgcn_wmma_f32_16x16x32_bf16(false, qf[f], false, bk1, (short)0, s1, false, false);
    }
    float mnew[8], alpha[8];
    #pragma unroll
    for (int v = 0; v < 8; ++v) {
      float a = fmaxf(s0[v], s1[v]) * 0.125f;
      #pragma unroll
      for (int off = 1; off < 16; off <<= 1) a = fmaxf(a, __shfl_xor(a, off, 32));
      mnew[v]  = fmaxf(mrow[v], a);
      alpha[v] = __expf(mrow[v] - mnew[v]);
    }
    #pragma unroll
    for (int v = 0; v < 8; ++v) {
      float p0 = __expf(s0[v] * 0.125f - mnew[v]);
      float p1 = __expf(s1[v] * 0.125f - mnew[v]);
      pl[(v + 8 * hi) * 32 + lm]      = f2bf(p0);
      pl[(v + 8 * hi) * 32 + lm + 16] = f2bf(p1);
      float r = p0 + p1;
      #pragma unroll
      for (int off = 1; off < 16; off <<= 1) r += __shfl_xor(r, off, 32);
      lrow[v] = lrow[v] * alpha[v] + r;
      mrow[v] = mnew[v];
    }
    #pragma unroll
    for (int dc = 0; dc < 4; ++dc)
      #pragma unroll
      for (int v = 0; v < 8; ++v) o[dc][v] *= alpha[v];

    Frag16 pf;
    pf.h[0] = *(const bf16x8*)&pl[lm * 32 + 16 * hi];
    pf.h[1] = *(const bf16x8*)&pl[lm * 32 + 16 * hi + 8];
    #pragma unroll
    for (int dc = 0; dc < 4; ++dc) {
      bf16x16 vf = *(const bf16x16*)(vbase + (size_t)(dc * 16 + lm) * SEQ + kk + 16 * hi);
      o[dc] = __builtin_amdgcn_wmma_f32_16x16x32_bf16(false, pf.v, false, vf, (short)0, o[dc], false, false);
    }
  }
  #pragma unroll
  for (int dc = 0; dc < 4; ++dc)
    #pragma unroll
    for (int v = 0; v < 8; ++v) {
      const int tok = b * SEQ + qrow0 + v + 8 * hi;
      outb[(size_t)tok * H_DIM + h * HDIM + dc * 16 + lm] = f2bf(o[dc][v] / lrow[v]);
    }
}

// ---------------------------------------------------------------- MoE routing / epilogues
__global__ void zero_counts(int* c) { if (threadIdx.x < 32) c[threadIdx.x] = 0; }

__global__ __launch_bounds__(256)
void build_buckets(const int* __restrict__ pos, const int* __restrict__ modal,
                   int E, int useModal, int* __restrict__ counts, int* __restrict__ rowmap)
{
  const int t = blockIdx.x * 256 + threadIdx.x;
  if (t >= N_TOK) return;
  int key = ((unsigned)pos[t]) % (unsigned)E;
  if (useModal) key += modal[t] * E;
  const int slot = atomicAdd(&counts[key], 1);
  rowmap[(size_t)key * MAPSTRIDE + slot] = t;
}

__global__ __launch_bounds__(256)
void silu_mul(const float* __restrict__ gu, bf16* __restrict__ inter, int IE)
{
  const size_t i = (size_t)blockIdx.x * 256 + threadIdx.x;
  if (i >= (size_t)N_TOK * IE) return;
  const int row = (int)(i / IE), c = (int)(i % IE);
  const float g = gu[(size_t)row * 2 * IE + c];
  const float u = gu[(size_t)row * 2 * IE + IE + c];
  inter[(size_t)row * IE + c] = f2bf((g / (1.f + __expf(-g))) * u);
}

// ---------------------------------------------------------------- host
static inline void* carve(char*& p, size_t bytes) {
  void* r = p;
  p += (bytes + 255) & ~(size_t)255;
  return r;
}

extern "C" void kernel_launch(void* const* d_in, const int* in_sizes, int n_in,
                              void* d_out, int out_size, void* d_ws, size_t ws_size,
                              hipStream_t stream)
{
  (void)in_sizes; (void)n_in; (void)out_size; (void)ws_size;
  const float* x      = (const float*)d_in[0];
  const float* qkv_w  = (const float*)d_in[1];
  const float* qkv_b  = (const float*)d_in[2];
  const float* proj_w = (const float*)d_in[3];
  const float* proj_b = (const float*)d_in[4];
  const float* ln1_g  = (const float*)d_in[5];
  const float* ln1_b  = (const float*)d_in[6];
  const float* ln2_g  = (const float*)d_in[7];
  const float* ln2_b  = (const float*)d_in[8];
  const float* ln3_g  = (const float*)d_in[9];
  const float* ln3_b  = (const float*)d_in[10];
  const float* gen_gu = (const float*)d_in[11];
  const float* gen_dn = (const float*)d_in[12];
  const float* modGU[4] = { (const float*)d_in[13], (const float*)d_in[15],
                            (const float*)d_in[17], (const float*)d_in[19] };
  const float* modDN[4] = { (const float*)d_in[14], (const float*)d_in[16],
                            (const float*)d_in[18], (const float*)d_in[20] };
  const int* modality = (const int*)d_in[21];
  const int* posids   = (const int*)d_in[22];
  float* out = (float*)d_out;

  char* wp = (char*)d_ws;
  bf16*  ln_bf    = (bf16*) carve(wp, (size_t)N_TOK * H_DIM * 2);
  float* qkv_f32  = (float*)carve(wp, (size_t)N_TOK * 3 * H_DIM * 4);
  bf16*  q_bf     = (bf16*) carve(wp, (size_t)N_TOK * H_DIM * 2);
  bf16*  k_bf     = (bf16*) carve(wp, (size_t)N_TOK * H_DIM * 2);
  bf16*  vT_bf    = (bf16*) carve(wp, (size_t)N_TOK * H_DIM * 2);
  bf16*  attn_bf  = (bf16*) carve(wp, (size_t)N_TOK * H_DIM * 2);
  float* gu_f32   = (float*)carve(wp, (size_t)N_TOK * 2048 * 4);
  bf16*  inter_bf = (bf16*) carve(wp, (size_t)N_TOK * 1024 * 2);
  int*   counts   = (int*)  carve(wp, 32 * 4);
  int*   rowmap   = (int*)  carve(wp, (size_t)16 * MAPSTRIDE * 4);

  // ---- attention branch -------------------------------------------------
  layernorm_bf16<<<N_TOK, 256, 0, stream>>>(x, ln1_g, ln1_b, ln_bf);
  gemm_bf16_wmma<<<dim3(3 * H_DIM / TILE_N, N_TOK / TILE_M, 1), 256, 0, stream>>>(
      ln_bf, H_DIM, nullptr, nullptr, 0, N_TOK,
      qkv_w, 0, 3 * H_DIM, H_DIM, qkv_b,
      qkv_f32, nullptr, nullptr, 3 * H_DIM, 0);
  split_qkv<<<(int)(((size_t)N_TOK * 3 * H_DIM + 255) / 256), 256, 0, stream>>>(
      qkv_f32, q_bf, k_bf, vT_bf);
  attn_wmma<<<dim3(2 * NHEADS, SEQ / 128), 256, 0, stream>>>(q_bf, k_bf, vT_bf, attn_bf);
  gemm_bf16_wmma<<<dim3(H_DIM / TILE_N, N_TOK / TILE_M, 1), 256, 0, stream>>>(
      attn_bf, H_DIM, nullptr, nullptr, 0, N_TOK,
      proj_w, 0, H_DIM, H_DIM, proj_b,
      out, nullptr, x, H_DIM, 1);                         // out = x + attn

  // ---- generation MoE (E=8, IE=512) ------------------------------------
  zero_counts<<<1, 32, 0, stream>>>(counts);
  build_buckets<<<N_TOK / 256, 256, 0, stream>>>(posids, nullptr, 8, 0, counts, rowmap);
  layernorm_bf16<<<N_TOK, 256, 0, stream>>>(out, ln2_g, ln2_b, ln_bf);
  gemm_bf16_wmma<<<dim3(1024 / TILE_N, N_TOK / TILE_M, 8), 256, 0, stream>>>(
      ln_bf, H_DIM, rowmap, counts, 0, N_TOK,
      gen_gu, (size_t)H_DIM * 1024, 1024, H_DIM, nullptr,
      gu_f32, nullptr, nullptr, 1024, 0);
  silu_mul<<<(int)(((size_t)N_TOK * 512 + 255) / 256), 256, 0, stream>>>(gu_f32, inter_bf, 512);
  gemm_bf16_wmma<<<dim3(H_DIM / TILE_N, N_TOK / TILE_M, 8), 256, 0, stream>>>(
      inter_bf, 512, rowmap, counts, 0, N_TOK,
      gen_dn, (size_t)512 * H_DIM, H_DIM, 512, nullptr,
      out, nullptr, nullptr, H_DIM, 3);                   // out += mlp

  // ---- modality MoE (4 sets, E=4, IE=1024) ------------------------------
  zero_counts<<<1, 32, 0, stream>>>(counts);
  build_buckets<<<N_TOK / 256, 256, 0, stream>>>(posids, modality, 4, 1, counts, rowmap);
  layernorm_bf16<<<N_TOK, 256, 0, stream>>>(out, ln3_g, ln3_b, ln_bf);
  for (int m = 0; m < 4; ++m) {
    gemm_bf16_wmma<<<dim3(2048 / TILE_N, N_TOK / TILE_M, 4), 256, 0, stream>>>(
        ln_bf, H_DIM, rowmap, counts, m * 4, N_TOK,
        modGU[m], (size_t)H_DIM * 2048, 2048, H_DIM, nullptr,
        gu_f32, nullptr, nullptr, 2048, 0);
    silu_mul<<<(int)(((size_t)N_TOK * 1024 + 255) / 256), 256, 0, stream>>>(gu_f32, inter_bf, 1024);
    gemm_bf16_wmma<<<dim3(H_DIM / TILE_N, N_TOK / TILE_M, 4), 256, 0, stream>>>(
        inter_bf, 1024, rowmap, counts, m * 4, N_TOK,
        modDN[m], (size_t)1024 * H_DIM, H_DIM, 1024, nullptr,
        out, nullptr, nullptr, H_DIM, 3);                 // out += masked mlp
  }
}
